// Sender_63900523430237
// MI455X (gfx1250) — compile-verified, matrix operand order checked by reading, not run on hardware
//
#include <hip/hip_runtime.h>
#include <hip/hip_bf16.h>

typedef __attribute__((ext_vector_type(16))) __bf16 v16bf;
typedef __attribute__((ext_vector_type(8)))  float  v8f;

#define B_ 8192
#define A_ 128
#define H_ 256
#define V_ 64
#define L_ 20
#define HSTRIDE 312   // bf16/LDS h-row stride: 156 dwords = 28 mod 64 banks -> conflict-free b128 reads
#define CSTRIDE 260   // f32/LDS c-row stride: 4 mod 64 banks -> half-wave row groups hit disjoint banks

// workspace layout (bytes) -- all weight arrays pre-swizzled to WMMA fragment layout
#define WS_ATTRB 0u          // attr  A-frags: [512 rowtiles][4 kc][32 lanes][16] bf16 = 2 MB
#define WS_WAB   2097152u    // W_a   B-frags: [16 tn][4 kc][32][16] bf16 = 64 KB
#define WS_WHHB  2162688u    // W_hh  B-frags: [64 tn][8 kc][32][16] bf16 = 512 KB
#define WS_WVB   2686976u    // W_v   B-frags: [4 tn][8 kc][32][16] bf16 = 32 KB
#define WS_BSUM  2719744u    // b_ih + b_hh f32 = 4 KB

// smem layout (bytes)
#define SM_C      0u         // 64*260 f32  = 66560
#define SM_LOGITS 66560u     // 64*64  f32  = 16384
#define SM_H0     82944u     // 64*312 bf16 = 39936
#define SM_H1     122880u    // 64*312 bf16 = 39936
#define SM_CH     162816u    // 64 int      = 256
#define SM_BYTES  163072u    // <= 163840 so two workgroups co-reside per 320KB WGP

// Load a pre-swizzled 32-byte WMMA fragment: lane l reads [base + (tile*32+l)*16 .. +15],
// so one wave covers a contiguous, perfectly coalesced 1KB block per (tile) access.
__device__ __forceinline__ v16bf ldfrag(const __bf16* base, int tile, int lane) {
  const __bf16* p = base + ((size_t)tile * 32 + lane) * 16;
  union { uint4 u[2]; v16bf v; } t;
  t.u[0] = *reinterpret_cast<const uint4*>(p);
  t.u[1] = *reinterpret_cast<const uint4*>(p + 8);
  return t.v;
}

// Gather a WMMA A fragment from LDS h (row-major, HSTRIDE): elements 0..7 = K[o..o+7],
// elements 8..15 = K[o+16..o+23], per the 16x16x32 bf16 operand layout.
__device__ __forceinline__ v16bf ldbf16x16(const __bf16* p, int o0) {
  union { uint4 u[2]; v16bf v; } t;
  t.u[0] = *reinterpret_cast<const uint4*>(p + o0);
  t.u[1] = *reinterpret_cast<const uint4*>(p + o0 + 16);
  return t.v;
}

__device__ __forceinline__ float sigf(float x) { return 1.0f / (1.0f + __expf(-x)); }
__device__ __forceinline__ float tanh_f(float x) {
  float e = __expf(2.0f * x);
  return (e - 1.0f) / (e + 1.0f);
}

// One-time: convert fp32 -> bf16 AND scatter into WMMA fragment layout.
// Fragment element (tile=(tn,kc), lane, e) holds W[row = tn*16 + (lane&15)]
//                                            [K  = kc*32 + (lane>>4)*8 + (e<8 ? e : e+8)].
__global__ void sender_prep(const float* __restrict__ attr, const float* __restrict__ Wa,
                            const float* __restrict__ Whh, const float* __restrict__ Wv,
                            const float* __restrict__ bih, const float* __restrict__ bhh,
                            __bf16* __restrict__ attrb, __bf16* __restrict__ Wab,
                            __bf16* __restrict__ Whhb, __bf16* __restrict__ Wvb,
                            float* __restrict__ bsum) {
  int i = blockIdx.x * blockDim.x + threadIdx.x;
  int e    = i & 15;
  int lane = (i >> 4) & 31;
  int kadd = ((lane >> 4) << 3) + (e < 8 ? e : e + 8);
  int row16 = lane & 15;

  if (i < B_ * A_) {   // attr: [512 rowtiles][4 kc]
    int kc = (i >> 9) & 3, rt = i >> 11;
    attrb[i] = (__bf16)attr[(size_t)(rt * 16 + row16) * A_ + kc * 32 + kadd];
  }
  if (i < H_ * A_) {   // W_a: [16 tn][4 kc]
    int kc = (i >> 9) & 3, tn = i >> 11;
    Wab[i] = (__bf16)Wa[(size_t)(tn * 16 + row16) * A_ + kc * 32 + kadd];
  }
  if (i < 4 * H_ * H_) {  // W_hh: [64 tn][8 kc]
    int kc = (i >> 9) & 7, tn = i >> 12;
    Whhb[i] = (__bf16)Whh[(size_t)(tn * 16 + row16) * H_ + kc * 32 + kadd];
  }
  if (i < V_ * H_) {   // W_v: [4 tn][8 kc]
    int kc = (i >> 9) & 7, tn = i >> 12;
    Wvb[i] = (__bf16)Wv[(size_t)(tn * 16 + row16) * H_ + kc * 32 + kadd];
  }
  if (i < 4 * H_) bsum[i] = bih[i] + bhh[i];
}

extern "C" __global__ __launch_bounds__(128)
void sender_main(const float* __restrict__ ba, const float* __restrict__ Wih,
                 const float* __restrict__ bv,
                 const __bf16* __restrict__ attrb, const __bf16* __restrict__ Wab,
                 const __bf16* __restrict__ Whhb, const __bf16* __restrict__ Wvb,
                 const float* __restrict__ bsum, float* __restrict__ out) {
  extern __shared__ __align__(16) char smem[];
  float*  c_lds      = (float*)(smem + SM_C);
  float*  logits_lds = (float*)(smem + SM_LOGITS);
  __bf16* hbuf0      = (__bf16*)(smem + SM_H0);
  __bf16* hbuf1      = (__bf16*)(smem + SM_H1);
  int*    ch_lds     = (int*)(smem + SM_CH);

  const int tid   = threadIdx.x;
  const int lane  = tid & 31;
  const int wave  = tid >> 5;
  const int lcol  = lane & 15;        // N (or M for A operand) within tile
  const int half8 = (lane >> 4) * 8;  // K-half select / M-offset for D
  const int wrow  = wave * 16;        // row base for row-partitioned phases
  const int growb = blockIdx.x * 64;  // global batch row base of workgroup

  __bf16* hcur  = hbuf0;
  __bf16* hnext = hbuf1;

  // ---------------- h0 = attr @ Wa^T + ba ; c = 0  (row-partitioned) ----------------
  {
    const int art = blockIdx.x * 4 + wave;  // attr row-tile of this wave
#pragma unroll 1
    for (int nt = 0; nt < 16; ++nt) {
      const int n = nt * 16 + lcol;
      float bias = ba[n];
      v8f acc;
#pragma unroll
      for (int r = 0; r < 8; ++r) acc[r] = bias;
#pragma unroll
      for (int kc = 0; kc < 4; ++kc) {
        v16bf av  = ldfrag(attrb, art * 4 + kc, lane);
        v16bf bvv = ldfrag(Wab, nt * 4 + kc, lane);
        acc = __builtin_amdgcn_wmma_f32_16x16x32_bf16(false, av, false, bvv,
                                                      (short)0, acc, false, false);
      }
#pragma unroll
      for (int r = 0; r < 8; ++r) {
        int lr = wrow + r + half8;
        hcur[lr * HSTRIDE + nt * 16 + lcol] = (__bf16)acc[r];
        c_lds[lr * CSTRIDE + nt * 16 + lcol] = 0.0f;
      }
    }
  }
  __syncthreads();  // h/c now visible to all waves (LSTM step is column-partitioned)

  float slp = 0.0f, plp = 0.0f, ep = 1.0f;

  // it=0: initial LSTM step (x = zeros). it=1..20: sample step l=it-1, then LSTM with one-hot x.
  for (int it = 0; it <= L_; ++it) {
    const bool hasx = (it > 0);

    if (hasx) {
      const int l = it - 1;
      // ---- logits = h @ Wv^T + bv (row-partitioned: wave's 16 rows) ----
      const __bf16* hrow = hcur + (size_t)(wrow + lcol) * HSTRIDE;
#pragma unroll 1
      for (int t = 0; t < 4; ++t) {
        const int n = t * 16 + lcol;
        float bias = bv[n];
        v8f acc;
#pragma unroll
        for (int r = 0; r < 8; ++r) acc[r] = bias;
#pragma unroll 2
        for (int kc = 0; kc < 8; ++kc) {
          v16bf av  = ldbf16x16(hrow, kc * 32 + half8);
          v16bf bvv = ldfrag(Wvb, t * 8 + kc, lane);
          acc = __builtin_amdgcn_wmma_f32_16x16x32_bf16(false, av, false, bvv,
                                                        (short)0, acc, false, false);
        }
#pragma unroll
        for (int r = 0; r < 8; ++r)
          logits_lds[(wrow + r + half8) * V_ + n] = acc[r];
      }
      __syncthreads();

      // ---- softmax / entropy / categorical sample, one row per thread 0..63 ----
      if (tid < 64) {
        const float* lr = logits_lds + tid * V_;
        float mx = -3.0e38f;
        for (int v = 0; v < V_; ++v) mx = fmaxf(mx, lr[v]);
        float s = 0.0f;
        for (int v = 0; v < V_; ++v) s += __expf(lr[v] - mx);
        float logZ = mx + __logf(s);
        float plp_add = 0.0f;
        for (int v = 0; v < V_; ++v) {
          float lp = lr[v] - logZ;
          plp_add += __expf(lp) * lp;
        }
        plp += plp_add;
        // deterministic hash RNG (stand-in for jax.random.categorical)
        unsigned grow = (unsigned)(growb + tid);
        unsigned x = grow * 0x9E3779B9u + (unsigned)(l + 1) * 0x85EBCA77u + 0x6C62272Eu;
        x ^= x >> 16; x *= 0x7FEB352Du; x ^= x >> 15; x *= 0x846CA68Bu; x ^= x >> 16;
        float u = (float)(x >> 8) * (1.0f / 16777216.0f);
        int chv = V_ - 1;
        float cum = 0.0f;
        for (int v = 0; v < V_; ++v) {
          cum += __expf(lr[v] - logZ);
          if (u < cum) { chv = v; break; }
        }
        float lp = lr[chv] - logZ;
        slp += lp;
        ep  *= __expf(lp);
        ch_lds[tid] = chv;
        out[(size_t)grow * L_ + l] = (float)chv;
      }
      __syncthreads();
    }

    // ---- LSTM step (column-partitioned): wave owns gate columns [wave*64, wave*64+64).
    //      Each W_hh B fragment is reused across the 4 row-tiles of the workgroup's 64 rows,
    //      so W_hh streams from L2 once per WORKGROUP per step, fully coalesced. ----
    {
#pragma unroll 1
      for (int nt = 0; nt < 4; ++nt) {
        const int ctile = wave * 4 + nt;      // gate-column tile (0..15) owned by this wave
        const int ncb   = ctile * 16;
        v8f acc[4][4];                        // [row-tile][gate]
#pragma unroll
        for (int gi = 0; gi < 4; ++gi) {
          const int gn = gi * H_ + ncb + lcol;
          float bias = bsum[gn];
#pragma unroll
          for (int rt = 0; rt < 4; ++rt) {
#pragma unroll
            for (int r = 0; r < 8; ++r) {
              float w = 0.0f;
              if (hasx) w = Wih[(size_t)gn * V_ + ch_lds[rt * 16 + r + half8]];
              acc[rt][gi][r] = bias + w;
            }
          }
        }
#pragma unroll 1
        for (int kc = 0; kc < 8; ++kc) {
          v16bf bvv[4];
#pragma unroll
          for (int gi = 0; gi < 4; ++gi)
            bvv[gi] = ldfrag(Whhb, (gi * 16 + ctile) * 8 + kc, lane);
#pragma unroll
          for (int rt = 0; rt < 4; ++rt) {
            v16bf av = ldbf16x16(hcur + (size_t)(rt * 16 + lcol) * HSTRIDE, kc * 32 + half8);
#pragma unroll
            for (int gi = 0; gi < 4; ++gi)
              acc[rt][gi] = __builtin_amdgcn_wmma_f32_16x16x32_bf16(
                  false, av, false, bvv[gi], (short)0, acc[rt][gi], false, false);
          }
        }
#pragma unroll
        for (int rt = 0; rt < 4; ++rt) {
#pragma unroll
          for (int r = 0; r < 8; ++r) {
            int lr = rt * 16 + r + half8;
            int ci = lr * CSTRIDE + ncb + lcol;
            float iv = acc[rt][0][r], fv = acc[rt][1][r];
            float gv = acc[rt][2][r], ov = acc[rt][3][r];
            float co = c_lds[ci];
            float cn = sigf(fv) * co + sigf(iv) * tanh_f(gv);
            c_lds[ci] = cn;
            hnext[lr * HSTRIDE + ncb + lcol] = (__bf16)(sigf(ov) * tanh_f(cn));
          }
        }
      }
      __syncthreads();  // h_next columns from all waves visible before next step reads rows
      __bf16* t = hcur; hcur = hnext; hnext = t;
    }
  }

  if (tid < 64) {
    unsigned grow = (unsigned)(growb + tid);
    out[(size_t)B_ * L_ + grow]            = slp;
    out[(size_t)B_ * L_ + B_ + grow]       = plp;
    out[(size_t)B_ * L_ + 2 * B_ + grow]   = ep;
  }
}

extern "C" void kernel_launch(void* const* d_in, const int* in_sizes, int n_in,
                              void* d_out, int out_size, void* d_ws, size_t ws_size,
                              hipStream_t stream) {
  (void)in_sizes; (void)n_in; (void)out_size; (void)ws_size;
  const float* attr = (const float*)d_in[0];
  const float* Wa   = (const float*)d_in[1];
  const float* ba   = (const float*)d_in[2];
  const float* Wih  = (const float*)d_in[3];
  const float* Whh  = (const float*)d_in[4];
  const float* bih  = (const float*)d_in[5];
  const float* bhh  = (const float*)d_in[6];
  const float* Wv   = (const float*)d_in[7];
  const float* bv   = (const float*)d_in[8];

  char* ws = (char*)d_ws;
  __bf16* attrb = (__bf16*)(ws + WS_ATTRB);
  __bf16* Wab   = (__bf16*)(ws + WS_WAB);
  __bf16* Whhb  = (__bf16*)(ws + WS_WHHB);
  __bf16* Wvb   = (__bf16*)(ws + WS_WVB);
  float*  bsum  = (float*)(ws + WS_BSUM);

  const int prepN = B_ * A_;  // largest segment
  sender_prep<<<(prepN + 255) / 256, 256, 0, stream>>>(attr, Wa, Whh, Wv, bih, bhh,
                                                       attrb, Wab, Whhb, Wvb, bsum);
  sender_main<<<B_ / 64, 128, SM_BYTES, stream>>>(ba, Wih, bv, attrb, Wab, Whhb, Wvb,
                                                  bsum, (float*)d_out);
}
